// VariationalSeq2Seq_meta_1151051235839
// MI455X (gfx1250) — matrix operand order repaired and checked by reading, not compiled, not run on hardware
//
#include <hip/hip_runtime.h>
#include <hip/hip_bf16.h>

// ---------------- problem constants ----------------
#define K_B    512
#define K_D    8
#define K_X    64
#define K_DX   512
#define K_H    256
#define K_LAT  256
#define K_E    3
#define K_OL   24
#define K_TENC 168
#define K_TDEC 24
#define CHUNK  8     // timesteps per phase-1 batch (168 = 21*8, 24 = 3*8)

typedef __bf16 bf16_t;
typedef __attribute__((ext_vector_type(16))) __bf16 v16bf;
typedef __attribute__((ext_vector_type(8)))  float  v8f;
typedef __attribute__((ext_vector_type(4)))  unsigned v4u;
typedef __attribute__((ext_vector_type(8)))  int    v8i_t;
typedef __attribute__((ext_vector_type(4)))  int    v4i_t;

// ---------------- small helpers ----------------
__device__ __forceinline__ bf16_t f2bf(float f) {
  unsigned u; __builtin_memcpy(&u, &f, 4);
  unsigned r = u + 0x7FFFu + ((u >> 16) & 1u);      // round-to-nearest-even
  unsigned short h = (unsigned short)(r >> 16);
  bf16_t o; __builtin_memcpy(&o, &h, 2); return o;
}
__device__ __forceinline__ float bf2f(bf16_t b) {
  unsigned short h; __builtin_memcpy(&h, &b, 2);
  unsigned u = ((unsigned)h) << 16; float f; __builtin_memcpy(&f, &u, 4); return f;
}
__device__ __forceinline__ float sigm(float x) { return 1.f / (1.f + __expf(-x)); }

__device__ __forceinline__ v8f wmma_bf16(v16bf a, v16bf b, v8f c) {
  // D = A(16x32 bf16) * B(32x16 bf16) + C(16x16 f32)
  return __builtin_amdgcn_wmma_f32_16x16x32_bf16(false, a, false, b, (short)0, c, false, false);
}

// A fragment (16x32 bf16, row-major source, leading dim ld):
// lane L holds row M=L&15; K groups {0..7,16..23} (L<16) or {8..15,24..31} (L>=16)
__device__ __forceinline__ v16bf load_a_frag(const bf16_t* base, int ld, int k0, int lane) {
  int row  = lane & 15;
  int koff = (lane >> 4) << 3;              // 0 or 8
  const bf16_t* p = base + (size_t)row * ld + k0 + koff;
  v16bf a;
#pragma unroll
  for (int j = 0; j < 8; ++j) { a[j] = p[j]; a[8 + j] = p[16 + j]; }
  return a;
}
// B fragment (32x16 bf16) from weight stored row-major (N,K): B(k,n)=W[n,k]
// lane L holds col N=L&15; K groups {0..7,16..23} (L<16) or {8..15,24..31} (L>=16)
__device__ __forceinline__ v16bf load_b_frag(const bf16_t* w, int ldK, int n0, int k0, int lane) {
  int col  = lane & 15;
  int koff = (lane >> 4) << 3;              // 0 or 8
  const bf16_t* p = w + (size_t)(n0 + col) * ldK + k0 + koff;
  v16bf b;
#pragma unroll
  for (int j = 0; j < 8; ++j) { b[j] = p[j]; b[8 + j] = p[16 + j]; }
  return b;
}

// ---------------- utility kernels ----------------
__global__ void cvt_f32_bf16_kernel(const float* __restrict__ src, bf16_t* __restrict__ dst, int n) {
  int i = blockIdx.x * 256 + threadIdx.x;
  if (i < n) dst[i] = f2bf(src[i]);
}
__global__ void fill0_kernel(float* __restrict__ p, int n) {
  int i = blockIdx.x * 256 + threadIdx.x;
  if (i < n) p[i] = 0.f;
}

// ---------------- phase 1a: h1 = tanh(x*W1 + b1), bf16 out ----------------
// rows ordered ((e*CHUNK + t)*K_B + b), 512 cols
__global__ void metanet_h1_kernel(const float* __restrict__ xt, const float* __restrict__ xw,
                                  const float* __restrict__ xs,
                                  const float* __restrict__ W1t, const float* __restrict__ b1t,
                                  const float* __restrict__ W1w, const float* __restrict__ b1w,
                                  const float* __restrict__ W1s, const float* __restrict__ b1s,
                                  bf16_t* __restrict__ out, int t0, int T) {
  int idx = blockIdx.x * 256 + threadIdx.x;          // 3*CHUNK*K_B*512 total
  int k = idx & 511; int r = idx >> 9;
  int b = r & (K_B - 1); int t = (r / K_B) % CHUNK; int e = r / (K_B * CHUNK);
  const float* x  = (e == 0) ? xt  : ((e == 1) ? xw  : xs);
  const float* W1 = (e == 0) ? W1t : ((e == 1) ? W1w : W1s);
  const float* b1 = (e == 0) ? b1t : ((e == 1) ? b1w : b1s);
  float xv = x[(size_t)b * T + (t0 + t)];
  out[idx] = f2bf(tanhf(xv * W1[k] + b1[k]));
}

// ---------------- phase 1b: W = LN(tanh(h1 @ W2^T + b2)) ----------------
// one workgroup = 16-row strip x 512 cols; 8 waves x (4 N-tiles of 16); K=512.
// The A strip (16x512 bf16 = 16KB, contiguous) is staged into LDS by the
// Tensor Data Mover (TDM) once per workgroup; all 8 waves then consume it
// from LDS instead of 8x-redundant global loads.
__global__ __launch_bounds__(256)
void gemm_tanh_ln_kernel(const bf16_t* __restrict__ A,        // rows x 512 (bf16)
                         const bf16_t* __restrict__ W2all,    // 3 x (512 x 512) bf16, (n,k)
                         const float* __restrict__ b2_0, const float* __restrict__ b2_1,
                         const float* __restrict__ b2_2,
                         const float* __restrict__ lnw, const float* __restrict__ lnb,
                         float* __restrict__ out, int rowsPerE) {
  __shared__ bf16_t As[16 * 512];          // 16KB TDM landing pad
  __shared__ float tile[16 * 512];
  __shared__ float ps[256], ps2[256];
  __shared__ float red[32];

  int tile0 = blockIdx.x * 16;
  int e = tile0 / rowsPerE;                          // rowsPerE = CHUNK*K_B (multiple of 16)
  const bf16_t* W2 = W2all + (size_t)e * 512 * 512;
  const float*  b2 = (e == 0) ? b2_0 : ((e == 1) ? b2_1 : b2_2);

  int wave = threadIdx.x >> 5, lane = threadIdx.x & 31;
  const bf16_t* Arow = A + (size_t)tile0 * 512;

  // ---- TDM: stage A strip into LDS (issued by wave 0 only; EXEC ignored) ----
  if (threadIdx.x < 32) {
    unsigned lds_addr = (unsigned)(unsigned long long)(void*)&As[0];
    unsigned long long ga = (unsigned long long)(const void*)Arow;
    const unsigned ELEMS = 16u * 512u;               // 8192 bf16 elements, 1-D tile
    v4u g0 = {};
    g0[0] = 1u;                                      // count=1 (valid user descriptor)
    g0[1] = lds_addr;                                // D#.lds_addr (bytes)
    g0[2] = (unsigned)(ga & 0xFFFFFFFFu);            // global_addr[31:0]
    g0[3] = (unsigned)((ga >> 32) & 0x01FFFFFFu)     // global_addr[56:32]
          | (2u << 30);                              // type=2 ("image")
    v8i_t g1 = {};
    g1[0] = (int)(1u << 16);                         // data_size=1 -> 2 bytes; mask/flags=0
    g1[1] = (int)((ELEMS & 0xFFFFu) << 16);          // tensor_dim0[15:0]
    g1[2] = (int)(((ELEMS >> 16) & 0xFFFFu)          // tensor_dim0[31:16]
          | (1u << 16));                             // tensor_dim1 = 1
    g1[3] = (int)((ELEMS & 0xFFFFu) << 16);          // tile_dim0 = 8192 (tile_dim1=0)
    g1[4] = 0;                                       // tile_dim1=0, tile_dim2=0
    g1[5] = (int)ELEMS;                              // tensor_dim0_stride[31:0]
    g1[6] = 0;                                       // stride0 hi / stride1 lo
    g1[7] = 0;
    v4i_t gz4 = {};
    v8i_t gz8 = {};
    __builtin_amdgcn_tensor_load_to_lds(g0, g1, gz4, gz4, gz8, 0);   // 6-arg (clang-23) form
    __builtin_amdgcn_s_wait_tensorcnt(0);
  }
  __syncthreads();

  int n_base = wave * 64;
  v8f acc[4] = {};
  for (int k0 = 0; k0 < 512; k0 += 32) {
    if (k0 + 32 < 512)   // pull next W2 K-block toward L2/L0 (global_prefetch_b8)
      __builtin_prefetch(W2 + (size_t)(n_base + (lane & 15)) * 512 + k0 + 32, 0, 1);
    v16bf a = load_a_frag(As, 512, k0, lane);        // ds_load from TDM-staged LDS
#pragma unroll
    for (int nt = 0; nt < 4; ++nt) {
      v16bf b = load_b_frag(W2, 512, n_base + nt * 16, k0, lane);
      acc[nt] = wmma_bf16(a, b, acc[nt]);
    }
  }
  // bias + tanh -> LDS
#pragma unroll
  for (int nt = 0; nt < 4; ++nt) {
    int n = n_base + nt * 16 + (lane & 15);
    int mb = (lane >> 4) * 8;
#pragma unroll
    for (int r = 0; r < 8; ++r)
      tile[(mb + r) * 512 + n] = tanhf(acc[nt][r] + b2[n]);
  }
  __syncthreads();

  // row LayerNorm over 512: 16 threads per row, each sums 32
  {
    int row = threadIdx.x >> 4, seg = threadIdx.x & 15;
    float s = 0.f, s2 = 0.f;
    const float* rp = &tile[row * 512 + seg * 32];
#pragma unroll 8
    for (int j = 0; j < 32; ++j) { float v = rp[j]; s += v; s2 += v * v; }
    ps[threadIdx.x] = s; ps2[threadIdx.x] = s2;
  }
  __syncthreads();
  if ((threadIdx.x & 15) == 0) {
    int row = threadIdx.x >> 4;
    float a = 0.f, b = 0.f;
    for (int j = 0; j < 16; ++j) { a += ps[row * 16 + j]; b += ps2[row * 16 + j]; }
    float m = a * (1.f / 512.f);
    float v = b * (1.f / 512.f) - m * m;
    red[row * 2] = m; red[row * 2 + 1] = rsqrtf(v + 1e-5f);
  }
  __syncthreads();
  float* orow = out + (size_t)tile0 * 512;
  for (int idx = threadIdx.x; idx < 16 * 512; idx += 256) {
    int r = idx >> 9, c = idx & 511;
    orow[(size_t)r * 512 + c] = (tile[idx] - red[r * 2]) * red[r * 2 + 1] * lnw[c] + lnb[c];
  }
}

// ---------------- phase 2: fused per-timestep step ----------------
// grid = 32 workgroups x 256 threads; each owns 16 batch rows
__global__ __launch_bounds__(256)
void step_kernel(float* __restrict__ hbuf,                     // (K_B,256) state, in/out
                 const float* __restrict__ Wt, int Tc, int t_local,  // chunk W tensors
                 const float* __restrict__ xl, int t_glob, int T,    // (B,T,D)
                 const float* __restrict__ theta0,             // 512
                 const bf16_t* __restrict__ gW1, const float* __restrict__ gb1,
                 const float* __restrict__ gW2, const float* __restrict__ gb2,
                 const bf16_t* __restrict__ Wih, const float* __restrict__ bih,
                 const bf16_t* __restrict__ Whh, const float* __restrict__ bhh,
                 int do_heads,
                 const float* __restrict__ hmW, const float* __restrict__ hmb,
                 const float* __restrict__ hlW, const float* __restrict__ hlb,
                 float* __restrict__ out_mu, float* __restrict__ out_lv, int tout) {
  __shared__ float  h_f[16 * 256];
  __shared__ bf16_t h_b[16 * 256];
  __shared__ bf16_t g1_b[16 * 256];
  __shared__ bf16_t xp_b[16 * 64];
  __shared__ float  gvals[16 * 3];

  int b0 = blockIdx.x * 16;
  int tid = threadIdx.x, wave = tid >> 5, lane = tid & 31;

  // stage h into LDS (f32 + bf16)
  for (int i = tid; i < 16 * 256; i += 256) {
    float v = hbuf[(size_t)(b0 + (i >> 8)) * 256 + (i & 255)];
    h_f[i] = v; h_b[i] = f2bf(v);
  }
  __syncthreads();

  // gating GEMM1: (16x256) @ (256x256)^T -> tanh -> bf16 LDS; wave owns 32 cols
  {
    int nb = wave * 32;
    v8f acc[2] = {};
    for (int k0 = 0; k0 < 256; k0 += 32) {
      v16bf a = load_a_frag(h_b, 256, k0, lane);
      acc[0] = wmma_bf16(a, load_b_frag(gW1, 256, nb,      k0, lane), acc[0]);
      acc[1] = wmma_bf16(a, load_b_frag(gW1, 256, nb + 16, k0, lane), acc[1]);
    }
#pragma unroll
    for (int nt = 0; nt < 2; ++nt) {
      int n = nb + nt * 16 + (lane & 15);
      int mb = (lane >> 4) * 8;
#pragma unroll
      for (int r = 0; r < 8; ++r)
        g1_b[(mb + r) * 256 + n] = f2bf(tanhf(acc[nt][r] + gb1[n]));
    }
  }
  __syncthreads();

  // logits (16x3) then top-2 masked softmax over E=3
  if (tid < 48) {
    int row = tid / 3, e = tid % 3;
    float s = gb2[e];
    for (int k = 0; k < 256; ++k) s += bf2f(g1_b[row * 256 + k]) * gW2[e * 256 + k];
    gvals[row * 3 + e] = s;
  }
  __syncthreads();
  if (tid < 16) {
    float l0 = gvals[tid * 3], l1 = gvals[tid * 3 + 1], l2 = gvals[tid * 3 + 2];
    int drop = (l0 <= l1 && l0 <= l2) ? 0 : ((l1 <= l2) ? 1 : 2);   // drop smallest (keep top-2)
    float m = fmaxf(fmaxf(l0, l1), l2);
    float e0 = (drop == 0) ? 0.f : __expf(l0 - m);
    float e1 = (drop == 1) ? 0.f : __expf(l1 - m);
    float e2 = (drop == 2) ? 0.f : __expf(l2 - m);
    float inv = 1.f / (e0 + e1 + e2);
    gvals[tid * 3] = e0 * inv; gvals[tid * 3 + 1] = e1 * inv; gvals[tid * 3 + 2] = e2 * inv;
  }
  __syncthreads();

  // theta combine + x' = einsum("bd,bdx->bx"): 16x64 outputs
  {
    size_t slab = (size_t)Tc * K_B * 512;
    for (int i = tid; i < 16 * 64; i += 256) {
      int row = i >> 6, x = i & 63;
      int b = b0 + row;
      float g0 = gvals[row * 3], g1 = gvals[row * 3 + 1], g2 = gvals[row * 3 + 2];
      float acc = 0.f;
#pragma unroll
      for (int d = 0; d < K_D; ++d) {
        int c = d * 64 + x;
        size_t base = ((size_t)t_local * K_B + b) * 512 + c;
        float th = g0 * Wt[base] + g1 * Wt[slab + base] + g2 * Wt[2 * slab + base] + theta0[c];
        acc += xl[((size_t)b * T + t_glob) * K_D + d] * th;
      }
      xp_b[i] = f2bf(acc);
    }
  }
  __syncthreads();

  // GRU: gi = x'(16x64)@Wih^T, gh = h(16x256)@Whh^T, N=768 split as r|z|n.
  // wave owns 32 cols of EACH gate so r/z/n meet in-register per (m,c).
  {
    int nb = wave * 32;
    v8f ai[6] = {};   // [gate*2 + nt]
    v8f ah[6] = {};
    for (int k0 = 0; k0 < 64; k0 += 32) {
      v16bf a = load_a_frag(xp_b, 64, k0, lane);
#pragma unroll
      for (int g = 0; g < 3; ++g)
#pragma unroll
        for (int nt = 0; nt < 2; ++nt)
          ai[g * 2 + nt] = wmma_bf16(a, load_b_frag(Wih, 64, g * 256 + nb + nt * 16, k0, lane), ai[g * 2 + nt]);
    }
    for (int k0 = 0; k0 < 256; k0 += 32) {
      v16bf a = load_a_frag(h_b, 256, k0, lane);
#pragma unroll
      for (int g = 0; g < 3; ++g)
#pragma unroll
        for (int nt = 0; nt < 2; ++nt)
          ah[g * 2 + nt] = wmma_bf16(a, load_b_frag(Whh, 256, g * 256 + nb + nt * 16, k0, lane), ah[g * 2 + nt]);
    }
    // fuse gates entirely in registers
#pragma unroll
    for (int nt = 0; nt < 2; ++nt) {
      int c = nb + nt * 16 + (lane & 15);
      int mb = (lane >> 4) * 8;
#pragma unroll
      for (int r = 0; r < 8; ++r) {
        int m = mb + r;
        float rr = sigm((ai[0 + nt][r] + bih[c])       + (ah[0 + nt][r] + bhh[c]));
        float zz = sigm((ai[2 + nt][r] + bih[256 + c]) + (ah[2 + nt][r] + bhh[256 + c]));
        float nn = tanhf((ai[4 + nt][r] + bih[512 + c]) + rr * (ah[4 + nt][r] + bhh[512 + c]));
        float hn = (1.f - zz) * nn + zz * h_f[m * 256 + c];
        h_f[m * 256 + c] = hn;
        hbuf[(size_t)(b0 + m) * 256 + c] = hn;
      }
    }
  }
  __syncthreads();

  // decoder heads: (16x256) @ (24x256)^T, both heads
  if (do_heads) {
    for (int i = tid; i < 16 * K_OL * 2; i += 256) {
      int which = (i >= 16 * K_OL);
      int j = i - which * 16 * K_OL;
      int row = j / K_OL, o = j % K_OL;
      const float* W = which ? hlW : hmW;
      const float* bb = which ? hlb : hmb;
      float s = bb[o];
      for (int k = 0; k < 256; ++k) s += h_f[row * 256 + k] * W[o * 256 + k];
      float* out = which ? out_lv : out_mu;
      out[((size_t)(b0 + row) * 25 + tout) * K_OL + o] = s;
    }
  }
}

// ---------------- latent: mu/logvar + reparameterize ----------------
__global__ void latent_kernel(const float* __restrict__ h,
                              const float* __restrict__ muW, const float* __restrict__ mub,
                              const float* __restrict__ lvW, const float* __restrict__ lvb,
                              const float* __restrict__ eps,
                              float* __restrict__ out_mu, float* __restrict__ out_lv,
                              float* __restrict__ z) {
  int b = blockIdx.x, c = threadIdx.x;   // 256 threads
  float sm = mub[c], sl = lvb[c];
  const float* hr = h + (size_t)b * 256;
  for (int k = 0; k < 256; ++k) { float hv = hr[k]; sm += hv * muW[c * 256 + k]; sl += hv * lvW[c * 256 + k]; }
  out_mu[(size_t)b * 256 + c] = sm;
  out_lv[(size_t)b * 256 + c] = sl;
  z[(size_t)b * 256 + c] = sm + eps[(size_t)b * 256 + c] * __expf(0.5f * sl);
}

// ---------------- heads on z (t=0 slot) ----------------
__global__ void heads0_kernel(const float* __restrict__ z,
                              const float* __restrict__ hmW, const float* __restrict__ hmb,
                              const float* __restrict__ hlW, const float* __restrict__ hlb,
                              float* __restrict__ out_mu, float* __restrict__ out_lv) {
  int b = blockIdx.x, tid = threadIdx.x;   // 48 threads
  int which = (tid >= K_OL), o = tid - which * K_OL;
  const float* W = which ? hlW : hmW;
  const float* bb = which ? hlb : hmb;
  float s = bb[o];
  const float* zr = z + (size_t)b * 256;
  for (int k = 0; k < 256; ++k) s += zr[k] * W[o * 256 + k];
  (which ? out_lv : out_mu)[((size_t)b * 25) * K_OL + o] = s;
}

// ---------------- host launcher ----------------
extern "C" void kernel_launch(void* const* d_in, const int* in_sizes, int n_in,
                              void* d_out, int out_size, void* d_ws, size_t ws_size,
                              hipStream_t stream) {
  (void)in_sizes; (void)n_in; (void)out_size; (void)ws_size;
  // raw inputs
  const float* enc_l = (const float*)d_in[0];
  const float* enc_t = (const float*)d_in[1];
  const float* enc_w = (const float*)d_in[2];
  const float* enc_s = (const float*)d_in[3];
  const float* dec_l = (const float*)d_in[4];
  const float* dec_t = (const float*)d_in[5];
  const float* dec_w = (const float*)d_in[6];
  const float* dec_s = (const float*)d_in[7];
  const float* eps   = (const float*)d_in[8];
  // params (pytree leaves in dict order)
  auto P = [&](int i) { return (const float*)d_in[i]; };
  // enc_block: 9..27 ; dec_block: 28..46
  const float* encW1[3] = { P(9),  P(13), P(17) };
  const float* encB1[3] = { P(10), P(14), P(18) };
  const float* encW2[3] = { P(11), P(15), P(19) };
  const float* encB2[3] = { P(12), P(16), P(20) };
  const float *enc_gW1 = P(21), *enc_gb1 = P(22), *enc_gW2 = P(23), *enc_gb2 = P(24);
  const float *enc_lnw = P(25), *enc_lnb = P(26), *enc_th0 = P(27);
  const float* decW1[3] = { P(28), P(32), P(36) };
  const float* decB1[3] = { P(29), P(33), P(37) };
  const float* decW2[3] = { P(30), P(34), P(38) };
  const float* decB2[3] = { P(31), P(35), P(39) };
  const float *dec_gW1 = P(40), *dec_gb1 = P(41), *dec_gW2 = P(42), *dec_gb2 = P(43);
  const float *dec_lnw = P(44), *dec_lnb = P(45), *dec_th0 = P(46);
  const float *enc_Wih = P(47), *enc_Whh = P(48), *enc_bih = P(49), *enc_bhh = P(50);
  const float *dec_Wih = P(51), *dec_Whh = P(52), *dec_bih = P(53), *dec_bhh = P(54);
  const float *muW = P(55), *mub = P(56), *lvW = P(57), *lvb = P(58);
  const float *hmW = P(59), *hmb = P(60), *hlW = P(61), *hlb = P(62);

  // outputs: mu_preds (B,25,24), lv_preds (B,25,24), mu (B,256), logvar (B,256)
  float* out_mu = (float*)d_out;
  float* out_lv = out_mu + (size_t)K_B * 25 * K_OL;
  float* out_m  = out_lv + (size_t)K_B * 25 * K_OL;
  float* out_v  = out_m  + (size_t)K_B * 256;

  // workspace carve-up
  char* ws = (char*)d_ws;
  size_t off = 0;
  auto alloc = [&](size_t bytes) { void* p = ws + off; off = (off + bytes + 255) & ~(size_t)255; return p; };
  bf16_t* W2bf_enc = (bf16_t*)alloc((size_t)3 * 512 * 512 * 2);
  bf16_t* W2bf_dec = (bf16_t*)alloc((size_t)3 * 512 * 512 * 2);
  bf16_t* gW1bf_e  = (bf16_t*)alloc((size_t)256 * 256 * 2);
  bf16_t* gW1bf_d  = (bf16_t*)alloc((size_t)256 * 256 * 2);
  bf16_t* Wihbf_e  = (bf16_t*)alloc((size_t)768 * 64 * 2);
  bf16_t* Wihbf_d  = (bf16_t*)alloc((size_t)768 * 64 * 2);
  bf16_t* Whhbf_e  = (bf16_t*)alloc((size_t)768 * 256 * 2);
  bf16_t* Whhbf_d  = (bf16_t*)alloc((size_t)768 * 256 * 2);
  float*  hbuf     = (float*)alloc((size_t)K_B * 256 * 4);
  float*  zbuf     = (float*)alloc((size_t)K_B * 256 * 4);
  bf16_t* h1buf    = (bf16_t*)alloc((size_t)3 * CHUNK * K_B * 512 * 2);
  float*  Wtbuf    = (float*)alloc((size_t)3 * CHUNK * K_B * 512 * 4);

  // ---- weight conversions (once per call) ----
  auto cvt = [&](const float* s, bf16_t* d, int n) {
    cvt_f32_bf16_kernel<<<(n + 255) / 256, 256, 0, stream>>>(s, d, n);
  };
  for (int e = 0; e < 3; ++e) cvt(encW2[e], W2bf_enc + (size_t)e * 512 * 512, 512 * 512);
  for (int e = 0; e < 3; ++e) cvt(decW2[e], W2bf_dec + (size_t)e * 512 * 512, 512 * 512);
  cvt(enc_gW1, gW1bf_e, 256 * 256);  cvt(dec_gW1, gW1bf_d, 256 * 256);
  cvt(enc_Wih, Wihbf_e, 768 * 64);   cvt(dec_Wih, Wihbf_d, 768 * 64);
  cvt(enc_Whh, Whhbf_e, 768 * 256);  cvt(dec_Whh, Whhbf_d, 768 * 256);
  fill0_kernel<<<(K_B * 256 + 255) / 256, 256, 0, stream>>>(hbuf, K_B * 256);

  const int rowsPerE   = CHUNK * K_B;                       // 4096
  const int h1_blocks  = (3 * CHUNK * K_B * 512) / 256;     // 24576
  const int gemm_tiles = (3 * rowsPerE) / 16;               // 768

  // ---- encoder: 21 chunks of 8 timesteps ----
  for (int c = 0; c < K_TENC / CHUNK; ++c) {
    int t0 = c * CHUNK;
    metanet_h1_kernel<<<h1_blocks, 256, 0, stream>>>(
        enc_t, enc_w, enc_s,
        encW1[0], encB1[0], encW1[1], encB1[1], encW1[2], encB1[2],
        h1buf, t0, K_TENC);
    gemm_tanh_ln_kernel<<<gemm_tiles, 256, 0, stream>>>(
        h1buf, W2bf_enc, encB2[0], encB2[1], encB2[2],
        enc_lnw, enc_lnb, Wtbuf, rowsPerE);
    for (int tl = 0; tl < CHUNK; ++tl) {
      step_kernel<<<K_B / 16, 256, 0, stream>>>(
          hbuf, Wtbuf, CHUNK, tl, enc_l, t0 + tl, K_TENC, enc_th0,
          gW1bf_e, enc_gb1, enc_gW2, enc_gb2,
          Wihbf_e, enc_bih, Whhbf_e, enc_bhh,
          /*do_heads=*/0, hmW, hmb, hlW, hlb, out_mu, out_lv, 0);
    }
  }

  // ---- latent + t=0 heads ----
  latent_kernel<<<K_B, 256, 0, stream>>>(hbuf, muW, mub, lvW, lvb, eps, out_m, out_v, zbuf);
  heads0_kernel<<<K_B, 48, 0, stream>>>(zbuf, hmW, hmb, hlW, hlb, out_mu, out_lv);

  // ---- decoder: 3 chunks of 8 timesteps, hidden = zbuf ----
  for (int c = 0; c < K_TDEC / CHUNK; ++c) {
    int t0 = c * CHUNK;
    metanet_h1_kernel<<<h1_blocks, 256, 0, stream>>>(
        dec_t, dec_w, dec_s,
        decW1[0], decB1[0], decW1[1], decB1[1], decW1[2], decB1[2],
        h1buf, t0, K_TDEC);
    gemm_tanh_ln_kernel<<<gemm_tiles, 256, 0, stream>>>(
        h1buf, W2bf_dec, decB2[0], decB2[1], decB2[2],
        dec_lnw, dec_lnb, Wtbuf, rowsPerE);
    for (int tl = 0; tl < CHUNK; ++tl) {
      step_kernel<<<K_B / 16, 256, 0, stream>>>(
          zbuf, Wtbuf, CHUNK, tl, dec_l, t0 + tl, K_TDEC, dec_th0,
          gW1bf_d, dec_gb1, dec_gW2, dec_gb2,
          Wihbf_d, dec_bih, Whhbf_d, dec_bhh,
          /*do_heads=*/1, hmW, hmb, hlW, hlb, out_mu, out_lv, t0 + tl + 1);
    }
  }
}
